// HeteroGATEncoderLinearDropout_15805479649919
// MI455X (gfx1250) — matrix operand order repaired and checked by reading, not compiled, not run on hardware
//
#include <hip/hip_runtime.h>
#include <hip/hip_bf16.h>

// ---------------------------------------------------------------------------
// HeteroGATv2 encoder forward for MI455X (gfx1250, wave32, WMMA).
//
// Roofline: GEMMs total ~10 GFLOP (negligible at any WMMA rate); the edge
// stage moves ~3.5 GB of irregular traffic -> ~150us at 23.3 TB/s HBM, and
// the lin1 output (102MB) fits in the 192MB L2 so per-edge gathers mostly
// hit L2.  GEMMs use v_wmma_f32_16x16x32_bf16 with A tiles staged into LDS
// via global_load_async_to_lds_b128 (ASYNCcnt); edge stage uses float4
// gathers + global f32 atomics for segment softmax / scatter-add.
// ---------------------------------------------------------------------------

#define N_NODES  50000
#define F_IN     128
#define E_EDGES  800000
#define H1       8
#define C1       16
#define F_OUT    64
#define ETOT     (E_EDGES + N_NODES)
#define NEG_SLOPE 0.2f

typedef __attribute__((ext_vector_type(16))) __bf16 v16bf;
typedef __attribute__((ext_vector_type(8)))  float  v8f;

union FragAB { v16bf v; unsigned int u[8]; };
union FragC  { v8f  v; float f[8]; };

// ------------------------------ helpers ------------------------------------

__global__ void f32_to_bf16_kernel(const float* __restrict__ src,
                                   __bf16* __restrict__ dst, long n) {
  long i = (long)blockIdx.x * blockDim.x + threadIdx.x;
  if (i < n) dst[i] = (__bf16)src[i];
}

__global__ void copy_f32_kernel(const float* __restrict__ src,
                                float* __restrict__ dst, long n) {
  long i = (long)blockIdx.x * blockDim.x + threadIdx.x;
  if (i < n) dst[i] = src[i];
}

__global__ void fill_f32_kernel(float* __restrict__ p, long n, float v) {
  long i = (long)blockIdx.x * blockDim.x + threadIdx.x;
  if (i < n) p[i] = v;
}

__device__ __forceinline__ void atomicMaxFloat(float* addr, float val) {
  unsigned int* ua = (unsigned int*)addr;
  unsigned int old = *ua;
  while (__uint_as_float(old) < val) {
    unsigned int assumed = old;
    old = atomicCAS(ua, assumed, __float_as_uint(val));
    if (old == assumed) break;
  }
}

// ------------------------------ WMMA GEMM ----------------------------------
// C[M x NC] = A[M x K] * W[NC x K]^T + bias.   M % 16 == 0, NC % 64 == 0,
// K % 32 == 0.  One 128-thread block = 4 waves; each wave owns one 16x16
// output tile; the 16 x K bf16 A tile is staged to LDS with the CDNA5
// async-to-LDS DMA path (global_load_async_to_lds_b128, ASYNCcnt) and
// shared by the 4 waves.  Fragment layouts per CDNA5 ISA 7.12.2:
//   A (16x32 bf16): lane m=l%16, half=l/16; vgpr v<4 -> K=2v+8*half(+0/1),
//                   v>=4 -> K=16+2(v-4)+8*half(+0/1)
//   B (32x16 bf16): lane col n=l%16, half=l/16; same K mapping as A
//   C (16x16 f32):  vgpr r -> row M=r+8*half, col N=l%16
template <int K, int NC>
__global__ __launch_bounds__(128)
void gemm_bf16_wmma(const __bf16* __restrict__ A, const __bf16* __restrict__ W,
                    const float* __restrict__ bias, float* __restrict__ C) {
  __shared__ __bf16 As[16 * K];
  const int tid = threadIdx.x;
  const long mTile = blockIdx.x;

  // stage A tile (rows mTile*16 .. +15 are one contiguous 32*K-byte block)
  // directly into LDS: flat-pointer low 32 bits == wave-relative LDS offset.
  {
    const unsigned long long gbase =
        (unsigned long long)(const void*)(A + (size_t)mTile * 16 * K);
    const unsigned int lbase = (unsigned int)(size_t)(void*)As;
#pragma unroll
    for (int i = tid; i < 2 * K; i += 128) {   // 2*K b128 transfers per tile
      unsigned int la = lbase + (unsigned int)i * 16u;
      unsigned long long ga = gbase + (unsigned long long)i * 16ull;
      asm volatile("global_load_async_to_lds_b128 %0, %1, off"
                   :: "v"(la), "v"(ga) : "memory");
    }
    asm volatile("s_wait_asynccnt 0" ::: "memory");
  }
  __syncthreads();

  const int wave = tid >> 5;
  const int lane = tid & 31;
  const int half = lane >> 4;
  const int ln   = lane & 15;
  const int nG   = (blockIdx.y * 4 + wave) * 16 + ln;
  const unsigned int* Wrow = (const unsigned int*)(W + (size_t)nG * K);

  FragC acc;
#pragma unroll
  for (int r = 0; r < 8; ++r) acc.f[r] = 0.f;

#pragma unroll
  for (int kk = 0; kk < K; kk += 32) {
    FragAB a, b;
#pragma unroll
    for (int v = 0; v < 8; ++v) {
      const int ka = kk + ((v < 4) ? 2 * v : 16 + 2 * (v - 4)) + 8 * half;
      a.u[v] = *(const unsigned int*)(As + ln * K + ka);
      const int kb = kk + ((v < 4) ? 2 * v : 16 + 2 * (v - 4)) + 8 * half;
      b.u[v] = Wrow[kb >> 1];
    }
    acc.v = __builtin_amdgcn_wmma_f32_16x16x32_bf16(
        false, a.v, false, b.v, (short)0, acc.v, false, false);
  }

  const float bv = bias ? bias[nG] : 0.f;
  const size_t rowBase = (size_t)mTile * 16 + 8 * half;
#pragma unroll
  for (int r = 0; r < 8; ++r)
    C[(rowBase + r) * (size_t)NC + nG] = acc.f[r] + bv;
}

// ------------------------------ edge stage ---------------------------------
// one thread per (edge, head).  Edges e >= E are the appended self loops.
// All row bases are 16-float aligned (CH, ldx multiples of 16; ws regions
// 256B-aligned), so gathers are float4.

__global__ void edge_score_kernel(const int* __restrict__ esrc,
                                  const int* __restrict__ edst, int E, int Nn,
                                  const float* __restrict__ xl,
                                  const float* __restrict__ xr, int ldx,
                                  const float* __restrict__ att, int H_, int CH,
                                  float* __restrict__ score,
                                  float* __restrict__ smax) {
  long tid = (long)blockIdx.x * blockDim.x + threadIdx.x;
  long total = (long)(E + Nn) * H_;
  if (tid >= total) return;
  int h = (int)(tid % H_);
  long e = tid / H_;
  int s, dn;
  if (e < E) { s = esrc[e]; dn = edst[e]; } else { s = dn = (int)(e - E); }
  const float4* pl = (const float4*)(xl + (size_t)s * ldx + h * CH);
  const float4* pr = (const float4*)(xr + (size_t)dn * ldx + h * CH);
  const float4* pa = (const float4*)(att + h * CH);
  float sc = 0.f;
  for (int c = 0; c < CH / 4; ++c) {
    float4 a = pl[c], b = pr[c], w = pa[c];
    float v0 = a.x + b.x; v0 = v0 > 0.f ? v0 : NEG_SLOPE * v0;
    float v1 = a.y + b.y; v1 = v1 > 0.f ? v1 : NEG_SLOPE * v1;
    float v2 = a.z + b.z; v2 = v2 > 0.f ? v2 : NEG_SLOPE * v2;
    float v3 = a.w + b.w; v3 = v3 > 0.f ? v3 : NEG_SLOPE * v3;
    sc += w.x * v0 + w.y * v1 + w.z * v2 + w.w * v3;
  }
  score[tid] = sc;
  atomicMaxFloat(&smax[(size_t)dn * H_ + h], sc);
}

__global__ void edge_aggregate_kernel(const int* __restrict__ esrc,
                                      const int* __restrict__ edst, int E,
                                      int Nn, const float* __restrict__ xl,
                                      int ldx, const float* __restrict__ score,
                                      const float* __restrict__ smax,
                                      float* __restrict__ denom,
                                      float* __restrict__ agg, int H_, int CH) {
  long tid = (long)blockIdx.x * blockDim.x + threadIdx.x;
  long total = (long)(E + Nn) * H_;
  if (tid >= total) return;
  int h = (int)(tid % H_);
  long e = tid / H_;
  int s, dn;
  if (e < E) { s = esrc[e]; dn = edst[e]; } else { s = dn = (int)(e - E); }
  float ex = __expf(score[tid] - smax[(size_t)dn * H_ + h]);
  atomicAdd(&denom[(size_t)dn * H_ + h], ex);
  const float4* pl = (const float4*)(xl + (size_t)s * ldx + h * CH);
  float* pd = agg + ((size_t)dn * H_ + h) * CH;
  for (int c = 0; c < CH / 4; ++c) {
    float4 a = pl[c];
    atomicAdd(&pd[4 * c + 0], a.x * ex);
    atomicAdd(&pd[4 * c + 1], a.y * ex);
    atomicAdd(&pd[4 * c + 2], a.z * ex);
    atomicAdd(&pd[4 * c + 3], a.w * ex);
  }
}

// hout[n, f] += agg[n, f] / denom[n, h] + bias[f]
__global__ void finalize_accum_kernel(const float* __restrict__ agg,
                                      const float* __restrict__ denom,
                                      const float* __restrict__ bias,
                                      float* __restrict__ hout, int Nn, int H_,
                                      int CH) {
  const int F = H_ * CH;
  long tid = (long)blockIdx.x * blockDim.x + threadIdx.x;
  long total = (long)Nn * F;
  if (tid >= total) return;
  long n = tid / F;
  int f = (int)(tid % F);
  int h = f / CH;
  hout[tid] += agg[tid] / denom[(size_t)n * H_ + h] + bias[f];
}

__global__ void elu_to_bf16_kernel(const float* __restrict__ src,
                                   __bf16* __restrict__ dst, long n) {
  long i = (long)blockIdx.x * blockDim.x + threadIdx.x;
  if (i < n) {
    float v = src[i];
    v = v > 0.f ? v : (__expf(v) - 1.f);
    dst[i] = (__bf16)v;
  }
}

// ------------------------------ launch -------------------------------------

static inline unsigned nblk(long n) { return (unsigned)((n + 255) / 256); }

extern "C" void kernel_launch(void* const* d_in, const int* in_sizes, int n_in,
                              void* d_out, int out_size, void* d_ws,
                              size_t ws_size, hipStream_t stream) {
  (void)in_sizes; (void)n_in; (void)out_size; (void)ws_size;
  const float* x     = (const float*)d_in[0];
  const int*   ec    = (const int*)d_in[1];   // [2, E]
  const int*   er    = (const int*)d_in[2];
  const float* Wl1c  = (const float*)d_in[3];
  const float* bl1c  = (const float*)d_in[4];
  const float* Wr1c  = (const float*)d_in[5];
  const float* br1c  = (const float*)d_in[6];
  const float* att1c = (const float*)d_in[7];
  const float* bias1c= (const float*)d_in[8];
  const float* Wl2c  = (const float*)d_in[9];
  const float* bl2c  = (const float*)d_in[10];
  const float* Wr2c  = (const float*)d_in[11];
  const float* br2c  = (const float*)d_in[12];
  const float* att2c = (const float*)d_in[13];
  const float* bias2c= (const float*)d_in[14];
  const float* Wl1r  = (const float*)d_in[15];
  const float* bl1r  = (const float*)d_in[16];
  const float* Wr1r  = (const float*)d_in[17];
  const float* br1r  = (const float*)d_in[18];
  const float* att1r = (const float*)d_in[19];
  const float* bias1r= (const float*)d_in[20];
  const float* Wl2r  = (const float*)d_in[21];
  const float* bl2r  = (const float*)d_in[22];
  const float* Wr2r  = (const float*)d_in[23];
  const float* br2r  = (const float*)d_in[24];
  const float* att2r = (const float*)d_in[25];
  const float* bias2r= (const float*)d_in[26];
  const float* Wlin  = (const float*)d_in[27];
  const float* blin  = (const float*)d_in[28];

  // workspace layout (bytes, all 256-aligned; regions reused across phases)
  char* ws = (char*)d_ws;
  const size_t OFF_XBF   = 0;                                  // N*128 bf16
  const size_t OFF_LIN1  = OFF_XBF   + (size_t)N_NODES*128*2;  // N*512 f32 (reused as lin2 N*256)
  const size_t OFF_SCORE = OFF_LIN1  + (size_t)N_NODES*512*4;  // ETOT*8 f32
  const size_t OFF_SMAX  = OFF_SCORE + (size_t)ETOT*8*4;       // N*8 f32
  const size_t OFF_DEN   = OFF_SMAX  + (size_t)N_NODES*8*4;    // N*8 f32
  const size_t OFF_AGG   = OFF_DEN   + (size_t)N_NODES*8*4;    // N*128 f32
  const size_t OFF_H1    = OFF_AGG   + (size_t)N_NODES*128*4;  // N*128 f32 (reused as h2 N*64)
  const size_t OFF_H1BF  = OFF_H1    + (size_t)N_NODES*128*4;  // N*128 bf16 (reused as h2bf)
  const size_t OFF_WC1   = OFF_H1BF  + (size_t)N_NODES*128*2;  // 512*128 bf16
  const size_t OFF_WC2   = OFF_WC1   + (size_t)512*128*2;      // 256*128 bf16
  const size_t OFF_WLN   = OFF_WC2   + (size_t)256*128*2;      // 64*64 bf16
  const size_t OFF_B1    = OFF_WLN   + (size_t)64*64*2;        // 512 f32
  const size_t OFF_B2    = OFF_B1    + 512*4;                  // 256 f32

  __bf16* xbf  = (__bf16*)(ws + OFF_XBF);
  float*  lin1 = (float*)(ws + OFF_LIN1);
  float*  scre = (float*)(ws + OFF_SCORE);
  float*  smax = (float*)(ws + OFF_SMAX);
  float*  den  = (float*)(ws + OFF_DEN);
  float*  agg  = (float*)(ws + OFF_AGG);
  float*  h1   = (float*)(ws + OFF_H1);
  __bf16* h1bf = (__bf16*)(ws + OFF_H1BF);
  __bf16* wc1  = (__bf16*)(ws + OFF_WC1);
  __bf16* wc2  = (__bf16*)(ws + OFF_WC2);
  __bf16* wln  = (__bf16*)(ws + OFF_WLN);
  float*  b1   = (float*)(ws + OFF_B1);
  float*  b2   = (float*)(ws + OFF_B2);
  float*  lin2 = lin1;   // region reuse
  float*  h2   = h1;
  __bf16* h2bf = h1bf;

  // ---- precision conversion: activations & weights -> bf16 ----
  f32_to_bf16_kernel<<<nblk((long)N_NODES*128), 256, 0, stream>>>(x, xbf, (long)N_NODES*128);
  f32_to_bf16_kernel<<<nblk(16384), 256, 0, stream>>>(Wl1c, wc1 + 0*16384, 16384);
  f32_to_bf16_kernel<<<nblk(16384), 256, 0, stream>>>(Wr1c, wc1 + 1*16384, 16384);
  f32_to_bf16_kernel<<<nblk(16384), 256, 0, stream>>>(Wl1r, wc1 + 2*16384, 16384);
  f32_to_bf16_kernel<<<nblk(16384), 256, 0, stream>>>(Wr1r, wc1 + 3*16384, 16384);
  f32_to_bf16_kernel<<<nblk(8192), 256, 0, stream>>>(Wl2c, wc2 + 0*8192, 8192);
  f32_to_bf16_kernel<<<nblk(8192), 256, 0, stream>>>(Wr2c, wc2 + 1*8192, 8192);
  f32_to_bf16_kernel<<<nblk(8192), 256, 0, stream>>>(Wl2r, wc2 + 2*8192, 8192);
  f32_to_bf16_kernel<<<nblk(8192), 256, 0, stream>>>(Wr2r, wc2 + 3*8192, 8192);
  f32_to_bf16_kernel<<<nblk(4096), 256, 0, stream>>>(Wlin, wln, 4096);
  copy_f32_kernel<<<1, 256, 0, stream>>>(bl1c, b1 + 0,   128);
  copy_f32_kernel<<<1, 256, 0, stream>>>(br1c, b1 + 128, 128);
  copy_f32_kernel<<<1, 256, 0, stream>>>(bl1r, b1 + 256, 128);
  copy_f32_kernel<<<1, 256, 0, stream>>>(br1r, b1 + 384, 128);
  copy_f32_kernel<<<1, 256, 0, stream>>>(bl2c, b2 + 0,   64);
  copy_f32_kernel<<<1, 256, 0, stream>>>(br2c, b2 + 64,  64);
  copy_f32_kernel<<<1, 256, 0, stream>>>(bl2r, b2 + 128, 64);
  copy_f32_kernel<<<1, 256, 0, stream>>>(br2r, b2 + 192, 64);

  // ---- layer 1: fused WMMA GEMM for [xl_c | xr_c | xl_r | xr_r] ----
  gemm_bf16_wmma<128, 512><<<dim3(N_NODES/16, 8), 128, 0, stream>>>(xbf, wc1, b1, lin1);

  fill_f32_kernel<<<nblk((long)N_NODES*128), 256, 0, stream>>>(h1, (long)N_NODES*128, 0.f);
  for (int t = 0; t < 2; ++t) {
    const int* src = t ? er : ec;
    const int* dst = src + E_EDGES;
    const float* att  = t ? att1r  : att1c;
    const float* bias = t ? bias1r : bias1c;
    const float* xl = lin1 + (t ? 256 : 0);
    const float* xr = lin1 + (t ? 384 : 128);
    fill_f32_kernel<<<nblk((long)N_NODES*8), 256, 0, stream>>>(smax, (long)N_NODES*8, -3.0e38f);
    fill_f32_kernel<<<nblk((long)N_NODES*8), 256, 0, stream>>>(den,  (long)N_NODES*8, 0.f);
    fill_f32_kernel<<<nblk((long)N_NODES*128), 256, 0, stream>>>(agg, (long)N_NODES*128, 0.f);
    edge_score_kernel<<<nblk((long)ETOT*H1), 256, 0, stream>>>(
        src, dst, E_EDGES, N_NODES, xl, xr, 512, att, H1, C1, scre, smax);
    edge_aggregate_kernel<<<nblk((long)ETOT*H1), 256, 0, stream>>>(
        src, dst, E_EDGES, N_NODES, xl, 512, scre, smax, den, agg, H1, C1);
    finalize_accum_kernel<<<nblk((long)N_NODES*128), 256, 0, stream>>>(
        agg, den, bias, h1, N_NODES, H1, C1);
  }
  elu_to_bf16_kernel<<<nblk((long)N_NODES*128), 256, 0, stream>>>(h1, h1bf, (long)N_NODES*128);

  // ---- layer 2: fused WMMA GEMM for [xl2_c | xr2_c | xl2_r | xr2_r] ----
  gemm_bf16_wmma<128, 256><<<dim3(N_NODES/16, 4), 128, 0, stream>>>(h1bf, wc2, b2, lin2);

  fill_f32_kernel<<<nblk((long)N_NODES*64), 256, 0, stream>>>(h2, (long)N_NODES*64, 0.f);
  for (int t = 0; t < 2; ++t) {
    const int* src = t ? er : ec;
    const int* dst = src + E_EDGES;
    const float* att  = t ? att2r  : att2c;
    const float* bias = t ? bias2r : bias2c;
    const float* xl = lin2 + (t ? 128 : 0);
    const float* xr = lin2 + (t ? 192 : 64);
    fill_f32_kernel<<<nblk((long)N_NODES), 256, 0, stream>>>(smax, (long)N_NODES, -3.0e38f);
    fill_f32_kernel<<<nblk((long)N_NODES), 256, 0, stream>>>(den,  (long)N_NODES, 0.f);
    fill_f32_kernel<<<nblk((long)N_NODES*64), 256, 0, stream>>>(agg, (long)N_NODES*64, 0.f);
    edge_score_kernel<<<nblk((long)ETOT), 256, 0, stream>>>(
        src, dst, E_EDGES, N_NODES, xl, xr, 256, att, 1, F_OUT, scre, smax);
    edge_aggregate_kernel<<<nblk((long)ETOT), 256, 0, stream>>>(
        src, dst, E_EDGES, N_NODES, xl, 256, scre, smax, den, agg, 1, F_OUT);
    finalize_accum_kernel<<<nblk((long)N_NODES*64), 256, 0, stream>>>(
        agg, den, bias, h2, N_NODES, 1, F_OUT);
  }
  elu_to_bf16_kernel<<<nblk((long)N_NODES*64), 256, 0, stream>>>(h2, h2bf, (long)N_NODES*64);

  // ---- final linear: h2 @ Wlin^T + blin -> d_out ----
  gemm_bf16_wmma<64, 64><<<dim3(N_NODES/16, 1), 128, 0, stream>>>(h2bf, wln, blin, (float*)d_out);
}
